// GNN_29764123361492
// MI455X (gfx1250) — matrix operand order repaired and checked by reading, not compiled, not run on hardware
//
#include <hip/hip_runtime.h>

// ---------------------------------------------------------------------------
// Problem constants (match reference)
// ---------------------------------------------------------------------------
#define GDIM   256   // g_dim (input feature dim)
#define H1DIM  256   // RGCN output dim
#define HD2    256   // H * D2 (transformer output dim)
#define RREL   8     // number of relations
#define EPSBN  1e-5f

typedef __attribute__((ext_vector_type(16))) __bf16   v16bf;
typedef __attribute__((ext_vector_type(8)))  __bf16   v8bf;
typedef __attribute__((ext_vector_type(8)))  float    v8f;
typedef __attribute__((ext_vector_type(4)))  unsigned u32x4;
typedef __attribute__((ext_vector_type(8)))  unsigned u32x8;

// ---------------------------------------------------------------------------
// Helpers
// ---------------------------------------------------------------------------
static __device__ __forceinline__ void atomAddF(float* p, float v) {
    __hip_atomic_fetch_add(p, v, __ATOMIC_RELAXED, __HIP_MEMORY_SCOPE_AGENT);
}

static __device__ __forceinline__ float waveSum(float v) {
    #pragma unroll
    for (int o = 16; o > 0; o >>= 1) v += __shfl_xor(v, o, 32);
    return v;
}

// monotone float -> uint key for atomicMax-based segment max
static __device__ __forceinline__ unsigned fkey(float f) {
    unsigned b = __float_as_uint(f);
    return (b & 0x80000000u) ? ~b : (b | 0x80000000u);
}
static __device__ __forceinline__ float fdekey(unsigned k) {
    return (k & 0x80000000u) ? __uint_as_float(k ^ 0x80000000u)
                             : __uint_as_float(~k);
}

// ---------------------------------------------------------------------------
// Utility kernels
// ---------------------------------------------------------------------------
__global__ void k_zero_u32(unsigned* __restrict__ p, int n) {
    int i = blockIdx.x * blockDim.x + threadIdx.x;
    if (i < n) p[i] = 0u;
}

__global__ void k_f32_to_bf16(const float* __restrict__ in,
                              __bf16* __restrict__ out, int n) {
    int i = blockIdx.x * blockDim.x + threadIdx.x;
    if (i < n) out[i] = (__bf16)in[i];
}

// per-(dst,relation) edge counts for RGCN mean aggregation
__global__ void k_count(const int* __restrict__ dst, const int* __restrict__ et,
                        float* __restrict__ cnt, int E) {
    int i = blockIdx.x * blockDim.x + threadIdx.x;
    if (i < E) atomAddF(&cnt[(size_t)dst[i] * RREL + et[i]], 1.0f);
}

// ---------------------------------------------------------------------------
// WMMA bf16 GEMM, K = Nc = 256 fixed:  C[M,256] = A[M,256] @ B[256,256] (+bias)
//
// Block = 128 threads = 4 waves. Block computes a 16x256 strip of C; each
// wave owns a 16x64 sub-strip (4 accumulators, 4 WMMAs per K-step).
//
// The 16x256 bf16 A tile (8 KB) is staged into LDS once per block by the
// Tensor Data Mover (tensor_load_to_lds, 2-D descriptor, 2 SGPR groups),
// wave 0 issues it, waits on TENSORcnt, then a workgroup barrier publishes
// the tile; all waves read A fragments via ds_load_b128.
// ---------------------------------------------------------------------------
__global__ __launch_bounds__(128) void k_gemm256(const __bf16* __restrict__ A,
                                                 const __bf16* __restrict__ B,
                                                 const float* __restrict__ bias,
                                                 float* __restrict__ C,
                                                 int M) {
    __shared__ __bf16 lds_a[16 * 256];

    const int lane = threadIdx.x & 31;
    const int wave = threadIdx.x >> 5;
    const int rowBase = blockIdx.x * 16;
    const int colBase = wave * 64;
    const int l15 = lane & 15;
    const int hi  = lane >> 4;

    if (wave == 0) {
        // ---- Tensor DMA descriptor (D#), 2-D tile: 256 x 16 bf16 --------
        const unsigned long long ga =
            (unsigned long long)(const void*)(A + (size_t)rowBase * 256);
        const unsigned ldsOff = (unsigned)(size_t)(const void*)lds_a; // addr[31:0] = LDS offset

        u32x4 g0;
        g0[0] = 1u;                                   // count=1, user mode
        g0[1] = ldsOff;                               // lds_addr
        g0[2] = (unsigned)(ga & 0xFFFFFFFFu);         // global_addr[31:0]
        g0[3] = (unsigned)((ga >> 32) & 0x01FFFFFFu)  // global_addr[56:32]
              | (2u << 30);                           // type = 2 ("image")

        u32x8 g1;
        g1[0] = (1u << 16);                 // workgroup_mask=0, data_size=1 (2B)
        g1[1] = (256u << 16);               // tensor_dim0 = 256 (bits 79:48)
        g1[2] = ((unsigned)M << 16);        // tensor_dim1 = M  (bits 111:80)
        g1[3] = ((unsigned)M >> 16) | (256u << 16);  // tile_dim0 = 256
        g1[4] = 16u;                        // tile_dim1 = 16, tile_dim2 = 0
        g1[5] = 256u;                       // tensor_dim0_stride = 256
        g1[6] = 0u;
        g1[7] = 0u;

        asm volatile("tensor_load_to_lds %0, %1" :: "s"(g0), "s"(g1) : "memory");
        __builtin_amdgcn_s_wait_tensorcnt(0);
    }
    __syncthreads();

    // A fragment source in LDS: row l15, K chunks at hi*8 and 16+hi*8
    const __bf16* la = lds_a + l15 * 256 + hi * 8;
    // B fragment: lane column = l15 (per 16-col group); K rows hi*8+i and 16+hi*8+i
    const __bf16* bp = B + (size_t)(hi * 8) * 256 + colBase + l15;

    v8f c0 = {}, c1 = {}, c2 = {}, c3 = {};
    #pragma unroll
    for (int kk = 0; kk < 256; kk += 32) {
        v8bf a0 = *(const v8bf*)(la + kk);            // ds_load_b128
        v8bf a1 = *(const v8bf*)(la + kk + 16);
        v16bf av;
        #pragma unroll
        for (int i = 0; i < 8; ++i) { av[i] = a0[i]; av[i + 8] = a1[i]; }

        const __bf16* bk = bp + (size_t)kk * 256;     // immediate offsets below
        __builtin_prefetch(bk + 32 * 256, 0, 1);
        v16bf b0, b1, b2, b3;
        #pragma unroll
        for (int i = 0; i < 8; ++i) {
            b0[i] = bk[i * 256 +  0]; b0[i + 8] = bk[(16 + i) * 256 +  0];
            b1[i] = bk[i * 256 + 16]; b1[i + 8] = bk[(16 + i) * 256 + 16];
            b2[i] = bk[i * 256 + 32]; b2[i + 8] = bk[(16 + i) * 256 + 32];
            b3[i] = bk[i * 256 + 48]; b3[i + 8] = bk[(16 + i) * 256 + 48];
        }

        c0 = __builtin_amdgcn_wmma_f32_16x16x32_bf16(false, av, false, b0, (short)0, c0, false, false);
        c1 = __builtin_amdgcn_wmma_f32_16x16x32_bf16(false, av, false, b1, (short)0, c1, false, false);
        c2 = __builtin_amdgcn_wmma_f32_16x16x32_bf16(false, av, false, b2, (short)0, c2, false, false);
        c3 = __builtin_amdgcn_wmma_f32_16x16x32_bf16(false, av, false, b3, (short)0, c3, false, false);
    }

    // C/D layout: lane holds column (group*16 + l15), rows rowBase + hi*8 + i
    v8f acc[4] = { c0, c1, c2, c3 };
    #pragma unroll
    for (int n = 0; n < 4; ++n) {
        const int col = colBase + n * 16 + l15;
        const float bc = bias ? bias[col] : 0.0f;
        #pragma unroll
        for (int i = 0; i < 8; ++i) {
            C[(size_t)(rowBase + hi * 8 + i) * 256 + col] = acc[n][i] + bc;
        }
    }
}

// ---------------------------------------------------------------------------
// RGCN edge scatter for one relation: agg[dst] += H_r[src] / cnt(dst,rel)
// One wave per edge, 8 f32 columns per lane.
// ---------------------------------------------------------------------------
__global__ void k_rgcn_scatter(const int* __restrict__ src,
                               const int* __restrict__ dst,
                               const int* __restrict__ et,
                               const float* __restrict__ cnt,
                               const float* __restrict__ Hr,
                               float* __restrict__ agg,
                               int E, int rel) {
    const int e = blockIdx.x * 8 + (threadIdx.x >> 5);
    const int lane = threadIdx.x & 31;
    if (e >= E) return;
    if (et[e] != rel) return;
    const int s = src[e], d = dst[e];
    const float nrm = 1.0f / fmaxf(cnt[(size_t)d * RREL + rel], 1.0f);
    const float4* hp = (const float4*)(Hr + (size_t)s * H1DIM) + lane * 2;
    float4 a = hp[0], b = hp[1];
    float* ap = agg + (size_t)d * H1DIM + lane * 8;
    atomAddF(ap + 0, a.x * nrm); atomAddF(ap + 1, a.y * nrm);
    atomAddF(ap + 2, a.z * nrm); atomAddF(ap + 3, a.w * nrm);
    atomAddF(ap + 4, b.x * nrm); atomAddF(ap + 5, b.y * nrm);
    atomAddF(ap + 6, b.z * nrm); atomAddF(ap + 7, b.w * nrm);
}

// ---------------------------------------------------------------------------
// Attention logits: lg[e] = dot(q[dst], k[src]) / sqrt(D2); segment max via key
// ---------------------------------------------------------------------------
__global__ void k_logits(const int* __restrict__ src, const int* __restrict__ dst,
                         const float* __restrict__ q, const float* __restrict__ k,
                         float* __restrict__ lg, unsigned* __restrict__ mkey, int E) {
    const int e = blockIdx.x * 8 + (threadIdx.x >> 5);
    const int lane = threadIdx.x & 31;
    if (e >= E) return;
    const int s = src[e], d = dst[e];
    const float4* qp = (const float4*)(q + (size_t)d * HD2) + lane * 2;
    const float4* kp = (const float4*)(k + (size_t)s * HD2) + lane * 2;
    float4 q0 = qp[0], q1 = qp[1], k0 = kp[0], k1 = kp[1];
    float p = q0.x * k0.x + q0.y * k0.y + q0.z * k0.z + q0.w * k0.w +
              q1.x * k1.x + q1.y * k1.y + q1.z * k1.z + q1.w * k1.w;
    p = waveSum(p);
    if (lane == 0) {
        float l = p * 0.0625f;                 // 1/sqrt(256)
        lg[e] = l;
        atomicMax(&mkey[d], fkey(l));
    }
}

// ex[e] = exp(lg[e]-max); den[dst] += ex   (lg buffer rewritten in place)
__global__ void k_expden(const int* __restrict__ dst, float* __restrict__ lg,
                         const unsigned* __restrict__ mkey,
                         float* __restrict__ den, int E) {
    int i = blockIdx.x * blockDim.x + threadIdx.x;
    if (i >= E) return;
    int d = dst[i];
    float ex = __expf(lg[i] - fdekey(mkey[d]));
    lg[i] = ex;
    atomAddF(&den[d], ex);
}

// out[dst] += (ex[e]/den[dst]) * v[src]   (out already holds skip path)
__global__ void k_attn_scatter(const int* __restrict__ src, const int* __restrict__ dst,
                               const float* __restrict__ ex, const float* __restrict__ den,
                               const float* __restrict__ v, float* __restrict__ out, int E) {
    const int e = blockIdx.x * 8 + (threadIdx.x >> 5);
    const int lane = threadIdx.x & 31;
    if (e >= E) return;
    const int s = src[e], d = dst[e];
    const float alpha = ex[e] / fmaxf(den[d], 1e-16f);
    const float4* vp = (const float4*)(v + (size_t)s * HD2) + lane * 2;
    float4 a = vp[0], b = vp[1];
    float* op = out + (size_t)d * HD2 + lane * 8;
    atomAddF(op + 0, a.x * alpha); atomAddF(op + 1, a.y * alpha);
    atomAddF(op + 2, a.z * alpha); atomAddF(op + 3, a.w * alpha);
    atomAddF(op + 4, b.x * alpha); atomAddF(op + 5, b.y * alpha);
    atomAddF(op + 6, b.z * alpha); atomAddF(op + 7, b.w * alpha);
}

// ---------------------------------------------------------------------------
// BatchNorm stats + apply + LeakyReLU
// ---------------------------------------------------------------------------
__global__ void k_bn_stats(const float* __restrict__ y, float* __restrict__ sum,
                           float* __restrict__ sumsq, int N) {
    const int col = threadIdx.x;                    // 256 threads
    const int chunk = (N + gridDim.x - 1) / gridDim.x;
    const int r0 = blockIdx.x * chunk;
    const int r1 = min(N, r0 + chunk);
    float s = 0.f, s2 = 0.f;
    for (int r = r0; r < r1; ++r) {
        float x = y[(size_t)r * HD2 + col];
        s += x; s2 += x * x;
    }
    atomAddF(&sum[col], s);
    atomAddF(&sumsq[col], s2);
}

__global__ void k_bn_apply(float* __restrict__ y, const float* __restrict__ sum,
                           const float* __restrict__ sumsq,
                           const float* __restrict__ gamma, const float* __restrict__ beta,
                           int N) {
    int i = blockIdx.x * blockDim.x + threadIdx.x;
    if (i >= N * HD2) return;
    int col = i & (HD2 - 1);
    float invN = 1.0f / (float)N;
    float mu = sum[col] * invN;
    float var = sumsq[col] * invN - mu * mu;
    float x = y[i];
    x = gamma[col] * (x - mu) * rsqrtf(var + EPSBN) + beta[col];
    y[i] = (x > 0.f) ? x : 0.01f * x;
}

// ---------------------------------------------------------------------------
// Launcher
// ---------------------------------------------------------------------------
static inline size_t alignUp(size_t v, size_t a) { return (v + a - 1) & ~(a - 1); }

extern "C" void kernel_launch(void* const* d_in, const int* in_sizes, int n_in,
                              void* d_out, int out_size, void* d_ws, size_t ws_size,
                              hipStream_t stream) {
    const float* x      = (const float*)d_in[0];
    const int*   ei     = (const int*)  d_in[1];
    const int*   etype  = (const int*)  d_in[2];
    const float* Wrel   = (const float*)d_in[3];
    const float* Wroot  = (const float*)d_in[4];
    const float* b1     = (const float*)d_in[5];
    const float* Wq     = (const float*)d_in[6];
    const float* bq     = (const float*)d_in[7];
    const float* Wk     = (const float*)d_in[8];
    const float* bk     = (const float*)d_in[9];
    const float* Wv     = (const float*)d_in[10];
    const float* bv     = (const float*)d_in[11];
    const float* Wskip  = (const float*)d_in[12];
    const float* bskip  = (const float*)d_in[13];
    const float* gamma  = (const float*)d_in[14];
    const float* beta   = (const float*)d_in[15];
    float* out = (float*)d_out;

    const int N = in_sizes[0] / GDIM;
    const int E = in_sizes[2];
    const int* srcv = ei;
    const int* dstv = ei + E;

    // ---- workspace carve-out -------------------------------------------
    char* w = (char*)d_ws;
    size_t off = 0;
    auto take = [&](size_t bytes) -> char* {
        char* p = w + off;
        off = alignUp(off + bytes, 256);
        return p;
    };
    __bf16* xbf    = (__bf16*)take((size_t)N * GDIM * 2);
    __bf16* wrelbf = (__bf16*)take((size_t)RREL * GDIM * H1DIM * 2);
    __bf16* wrootbf= (__bf16*)take((size_t)GDIM * H1DIM * 2);
    __bf16* wqbf   = (__bf16*)take((size_t)H1DIM * HD2 * 2);
    __bf16* wkbf   = (__bf16*)take((size_t)H1DIM * HD2 * 2);
    __bf16* wvbf   = (__bf16*)take((size_t)H1DIM * HD2 * 2);
    __bf16* wsbf   = (__bf16*)take((size_t)H1DIM * HD2 * 2);
    float*  cnt    = (float*) take((size_t)N * RREL * 4);
    float*  Hr     = (float*) take((size_t)N * H1DIM * 4);
    float*  agg    = (float*) take((size_t)N * H1DIM * 4);
    __bf16* x1bf   = (__bf16*)take((size_t)N * H1DIM * 2);
    float*  qbuf   = (float*) take((size_t)N * HD2 * 4);
    float*  kbuf   = (float*) take((size_t)N * HD2 * 4);
    float*  vbuf   = (float*) take((size_t)N * HD2 * 4);
    float*  lgbuf  = (float*) take((size_t)E * 4);
    unsigned* mkey = (unsigned*)take((size_t)N * 4);
    float*  den    = (float*) take((size_t)N * 4);
    float*  bnsum  = (float*) take(HD2 * 4);
    float*  bnss   = (float*) take(HD2 * 4);
    (void)ws_size; (void)n_in; (void)out_size;

    const int T = 256;
    auto blocks = [&](long n) { return dim3((unsigned)((n + T - 1) / T)); };

    // ---- 0. zero accumulators ------------------------------------------
    k_zero_u32<<<blocks((long)N * RREL), T, 0, stream>>>((unsigned*)cnt, N * RREL);
    k_zero_u32<<<blocks(N), T, 0, stream>>>(mkey, N);
    k_zero_u32<<<blocks(N), T, 0, stream>>>((unsigned*)den, N);
    k_zero_u32<<<1, T, 0, stream>>>((unsigned*)bnsum, HD2);
    k_zero_u32<<<1, T, 0, stream>>>((unsigned*)bnss, HD2);

    // ---- 1. fp32 -> bf16 conversions -----------------------------------
    k_f32_to_bf16<<<blocks((long)N * GDIM), T, 0, stream>>>(x, xbf, N * GDIM);
    k_f32_to_bf16<<<blocks((long)RREL * GDIM * H1DIM), T, 0, stream>>>(Wrel, wrelbf, RREL * GDIM * H1DIM);
    k_f32_to_bf16<<<blocks((long)GDIM * H1DIM), T, 0, stream>>>(Wroot, wrootbf, GDIM * H1DIM);
    k_f32_to_bf16<<<blocks((long)H1DIM * HD2), T, 0, stream>>>(Wq, wqbf, H1DIM * HD2);
    k_f32_to_bf16<<<blocks((long)H1DIM * HD2), T, 0, stream>>>(Wk, wkbf, H1DIM * HD2);
    k_f32_to_bf16<<<blocks((long)H1DIM * HD2), T, 0, stream>>>(Wv, wvbf, H1DIM * HD2);
    k_f32_to_bf16<<<blocks((long)H1DIM * HD2), T, 0, stream>>>(Wskip, wsbf, H1DIM * HD2);

    // ---- 2. per-(dst,rel) edge counts ----------------------------------
    k_count<<<blocks(E), T, 0, stream>>>(dstv, etype, cnt, E);

    // ---- 3. RGCN: agg = x@Wroot + b1, then per-relation scatter --------
    dim3 gGemm((N + 15) / 16);             // one 16-row strip per block
    k_gemm256<<<gGemm, 128, 0, stream>>>(xbf, wrootbf, b1, agg, N);
    dim3 gEdge((E + 7) / 8);
    for (int r = 0; r < RREL; ++r) {
        k_gemm256<<<gGemm, 128, 0, stream>>>(xbf, wrelbf + (size_t)r * GDIM * H1DIM,
                                             nullptr, Hr, N);
        k_rgcn_scatter<<<gEdge, T, 0, stream>>>(srcv, dstv, etype, cnt, Hr, agg, E, r);
    }

    // ---- 4. x1 -> bf16, transformer linears ----------------------------
    k_f32_to_bf16<<<blocks((long)N * H1DIM), T, 0, stream>>>(agg, x1bf, N * H1DIM);
    k_gemm256<<<gGemm, 128, 0, stream>>>(x1bf, wqbf, bq, qbuf, N);
    k_gemm256<<<gGemm, 128, 0, stream>>>(x1bf, wkbf, bk, kbuf, N);
    k_gemm256<<<gGemm, 128, 0, stream>>>(x1bf, wvbf, bv, vbuf, N);
    k_gemm256<<<gGemm, 128, 0, stream>>>(x1bf, wsbf, bskip, out, N); // skip path

    // ---- 5. attention ---------------------------------------------------
    k_logits<<<gEdge, T, 0, stream>>>(srcv, dstv, qbuf, kbuf, lgbuf, mkey, E);
    k_expden<<<blocks(E), T, 0, stream>>>(dstv, lgbuf, mkey, den, E);
    k_attn_scatter<<<gEdge, T, 0, stream>>>(srcv, dstv, lgbuf, den, vbuf, out, E);

    // ---- 6. BatchNorm + LeakyReLU --------------------------------------
    k_bn_stats<<<dim3(128), T, 0, stream>>>(out, bnsum, bnss, N);
    k_bn_apply<<<blocks((long)N * HD2), T, 0, stream>>>(out, bnsum, bnss, gamma, beta, N);
}
